// MambaSSM_33784212750960
// MI455X (gfx1250) — compile-verified, matrix operand order checked by reading, not run on hardware
//
#include <hip/hip_runtime.h>
#include <hip/hip_bf16.h>
#include <math.h>

#define D_MODEL 256
#define D_INNER 512
#define D_STATE 16
#define KCONV   4
#define BATCH   8
#define TLEN    2048
#define ROWS    (BATCH * TLEN)     // 16384
#define NCHUNK  16
#define CHUNK   (TLEN / NCHUNK)    // 128
#define NPAD    48                 // 33 proj cols padded to 3 WMMA tiles

typedef __attribute__((ext_vector_type(16))) _Float16 v16h;
typedef __attribute__((ext_vector_type(8)))  float    v8f;

__device__ __forceinline__ float sigmoidf_(float x) { return 1.0f / (1.0f + __expf(-x)); }
__device__ __forceinline__ float softplusf_(float x) { return (x > 20.0f) ? x : log1pf(__expf(x)); }

// ---------------------------------------------------------------------------
// Pack a row-major f32 B[KxN] into fragment-ordered f16 so each lane of a wave
// can fetch its 16 B elements for one (k-step, n-tile) as a contiguous 32B load.
// Element e of lane L in k-step ks covers kk = ks*32 + (L/16)*8 + (e<8?e:e+8),
// column tn*16 + (L%16)  (ISA 7.12.2 16-bit B layout, symmetric to A).
// Bp index: ((tn*ksteps + ks)*32 + lane)*16 + e
// ---------------------------------------------------------------------------
__global__ void pack_b_f16(const float* __restrict__ B, _Float16* __restrict__ Bp,
                           int N, int K) {
    const int ksteps = K >> 5;
    const int total  = (N >> 4) * ksteps * 32;
    const int idx    = blockIdx.x * blockDim.x + threadIdx.x;
    if (idx >= total) return;
    const int lane = idx & 31;
    const int ks   = (idx >> 5) % ksteps;
    const int tn   = (idx >> 5) / ksteps;
    const int kb   = (ks << 5) + (lane >> 4) * 8;
    const int col  = tn * 16 + (lane & 15);
    _Float16* dst  = Bp + (size_t)idx * 16;
#pragma unroll
    for (int e = 0; e < 16; ++e) {
        const int kk = kb + ((e < 8) ? e : (e + 8));
        dst[e] = (_Float16)B[(size_t)kk * N + col];
    }
}

// ---------------------------------------------------------------------------
// Register-blocked GEMM: each wave owns an MT x NT grid of 16x16 tiles.
// A: f32 row-major (converted inline, contiguous b128 loads per the A-frag
// layout); B: pre-packed f16 fragments; C: f32 row-major.
// Per K-step: MT A-frags + NT B-frag loads feed MT*NT v_wmma_f32_16x16x32_f16.
// ---------------------------------------------------------------------------
template <int MT, int NT>
__global__ void gemm_wmma_packed(const float* __restrict__ A, const _Float16* __restrict__ Bp,
                                 float* __restrict__ C, int M, int N, int K) {
    const int ksteps  = K >> 5;
    const int groupsN = (N >> 4) / NT;
    const int groupsM = (M >> 4) / MT;
    const int wave    = (blockIdx.x * blockDim.x + threadIdx.x) >> 5;
    if (wave >= groupsM * groupsN) return;    // uniform per wave: EXEC all-ones
    const int lane  = threadIdx.x & 31;
    const int tm0   = (wave / groupsN) * MT;
    const int tn0   = (wave % groupsN) * NT;
    const int half  = lane >> 4;
    const int l16   = lane & 15;
    const int kbase = half * 8;
    const v16h* Bv  = (const v16h*)Bp;

    v8f acc[MT][NT];
#pragma unroll
    for (int i = 0; i < MT; ++i)
#pragma unroll
        for (int j = 0; j < NT; ++j) acc[i][j] = (v8f){};

    for (int ks = 0; ks < ksteps; ++ks) {
        const int k0 = ks << 5;
        v16h a[MT];
#pragma unroll
        for (int i = 0; i < MT; ++i) {
            const float* ap = A + (size_t)((tm0 + i) * 16 + l16) * K + k0 + kbase;
            __builtin_prefetch(ap + 32, 0, 1);            // next K-step row
            const float4 p0 = *(const float4*)(ap);
            const float4 p1 = *(const float4*)(ap + 4);
            const float4 p2 = *(const float4*)(ap + 16);
            const float4 p3 = *(const float4*)(ap + 20);
            a[i][0]  = (_Float16)p0.x;  a[i][1]  = (_Float16)p0.y;
            a[i][2]  = (_Float16)p0.z;  a[i][3]  = (_Float16)p0.w;
            a[i][4]  = (_Float16)p1.x;  a[i][5]  = (_Float16)p1.y;
            a[i][6]  = (_Float16)p1.z;  a[i][7]  = (_Float16)p1.w;
            a[i][8]  = (_Float16)p2.x;  a[i][9]  = (_Float16)p2.y;
            a[i][10] = (_Float16)p2.z;  a[i][11] = (_Float16)p2.w;
            a[i][12] = (_Float16)p3.x;  a[i][13] = (_Float16)p3.y;
            a[i][14] = (_Float16)p3.z;  a[i][15] = (_Float16)p3.w;
        }
        v16h b[NT];
#pragma unroll
        for (int j = 0; j < NT; ++j)
            b[j] = Bv[((size_t)(tn0 + j) * ksteps + ks) * 32 + lane];
#pragma unroll
        for (int i = 0; i < MT; ++i)
#pragma unroll
            for (int j = 0; j < NT; ++j)
                acc[i][j] = __builtin_amdgcn_wmma_f32_16x16x32_f16(
                    false, a[i], false, b[j], (short)0, acc[i][j], false, false);
    }

#pragma unroll
    for (int i = 0; i < MT; ++i)
#pragma unroll
        for (int r = 0; r < 8; ++r) {
            const int m = (tm0 + i) * 16 + half * 8 + r;
#pragma unroll
            for (int j = 0; j < NT; ++j)
                C[(size_t)m * N + (tn0 + j) * 16 + l16] = acc[i][j][r];
        }
}

// ---------------------------------------------------------------------------
// Depthwise causal conv (K=4) + bias + SiLU gate:  u = conv(x_part) * sigmoid(z_part)
// ---------------------------------------------------------------------------
__global__ void conv_gate(const float* __restrict__ xz, const float* __restrict__ cw,
                          const float* __restrict__ cb, float* __restrict__ u) {
    const int row = blockIdx.x;          // b*T + t
    const int d   = threadIdx.x;         // 0..511
    const int t   = row % TLEN;
    float acc = cb[d];
#pragma unroll
    for (int k = 0; k < KCONV; ++k) {
        const int ts = t - (KCONV - 1) + k;     // causal, zero-padded per sequence
        if (ts >= 0)
            acc += cw[d * KCONV + k] * xz[(size_t)(row + ts - t) * (2 * D_INNER) + d];
    }
    const float z = xz[(size_t)row * (2 * D_INNER) + D_INNER + d];
    u[(size_t)row * D_INNER + d] = acc * sigmoidf_(z);
}

// Pad W_xproj (512x33) -> (512x48) with zeros so GEMM2 is clean 16-wide tiles.
__global__ void pad_wxproj(const float* __restrict__ w, float* __restrict__ wp) {
    const int i = blockIdx.x * blockDim.x + threadIdx.x;
    if (i >= D_INNER * NPAD) return;
    const int r = i / NPAD, c = i % NPAD;
    wp[i] = (c < 1 + 2 * D_STATE) ? w[r * (1 + 2 * D_STATE) + c] : 0.0f;
}

// ---------------------------------------------------------------------------
// Chunked linear-recurrence scan, phase 1: per (b, d, chunk) local scan from a
// zero state; chunk decay product is exp(A[d,n] * sum(dt)). xp chunk in LDS.
// ---------------------------------------------------------------------------
__global__ void scan_partial(const float* __restrict__ xp, const float* __restrict__ u,
                             const float* __restrict__ w_dt, const float* __restrict__ b_dt,
                             const float* __restrict__ A,
                             float* __restrict__ P, float* __restrict__ S) {
    __shared__ float sxp[CHUNK * NPAD];
    const int blk = blockIdx.x;
    const int h   = blk & 1;
    const int c   = (blk >> 1) % NCHUNK;
    const int b   = blk / (2 * NCHUNK);
    const int d   = h * 256 + threadIdx.x;

    for (int i = threadIdx.x; i < CHUNK * NPAD; i += blockDim.x)
        sxp[i] = xp[(size_t)(b * TLEN + c * CHUNK) * NPAD + i];
    __syncthreads();

    float Ad[D_STATE];
#pragma unroll
    for (int n = 0; n < D_STATE; ++n) Ad[n] = A[d * D_STATE + n];
    const float wdt = w_dt[d], bdt = b_dt[d];

    float s[D_STATE];
#pragma unroll
    for (int n = 0; n < D_STATE; ++n) s[n] = 0.0f;
    float sumdt = 0.0f;

    for (int tt = 0; tt < CHUNK; ++tt) {
        const int   t    = c * CHUNK + tt;
        const float uval = u[(size_t)(b * TLEN + t) * D_INNER + d];
        const float dt   = softplusf_(sxp[tt * NPAD] * wdt + bdt);
        sumdt += dt;
        const float du = dt * uval;
#pragma unroll
        for (int n = 0; n < D_STATE; ++n)
            s[n] = s[n] * __expf(dt * Ad[n]) + du * sxp[tt * NPAD + 1 + n];
    }
    const size_t o = (((size_t)b * NCHUNK + c) * D_INNER + d) * D_STATE;
#pragma unroll
    for (int n = 0; n < D_STATE; ++n) {
        P[o + n] = __expf(Ad[n] * sumdt);
        S[o + n] = s[n];
    }
}

// Phase 2: sequential combine across the 16 chunks per (b,d) channel.
__global__ void scan_carry(const float* __restrict__ P, const float* __restrict__ S,
                           float* __restrict__ init) {
    const int idx = blockIdx.x * blockDim.x + threadIdx.x;
    if (idx >= BATCH * D_INNER) return;
    const int b = idx / D_INNER, d = idx % D_INNER;
    float carry[D_STATE];
#pragma unroll
    for (int n = 0; n < D_STATE; ++n) carry[n] = 0.0f;
    for (int ch = 0; ch < NCHUNK; ++ch) {
        const size_t o = (((size_t)b * NCHUNK + ch) * D_INNER + d) * D_STATE;
#pragma unroll
        for (int n = 0; n < D_STATE; ++n) {
            init[o + n] = carry[n];
            carry[n]    = carry[n] * P[o + n] + S[o + n];
        }
    }
}

// Phase 3: re-run each chunk from its correct initial state, emit y.
__global__ void scan_final(const float* __restrict__ xp, const float* __restrict__ u,
                           const float* __restrict__ w_dt, const float* __restrict__ b_dt,
                           const float* __restrict__ A, const float* __restrict__ Dp,
                           const float* __restrict__ init, float* __restrict__ y) {
    __shared__ float sxp[CHUNK * NPAD];
    const int blk = blockIdx.x;
    const int h   = blk & 1;
    const int c   = (blk >> 1) % NCHUNK;
    const int b   = blk / (2 * NCHUNK);
    const int d   = h * 256 + threadIdx.x;

    for (int i = threadIdx.x; i < CHUNK * NPAD; i += blockDim.x)
        sxp[i] = xp[(size_t)(b * TLEN + c * CHUNK) * NPAD + i];
    __syncthreads();

    float Ad[D_STATE];
#pragma unroll
    for (int n = 0; n < D_STATE; ++n) Ad[n] = A[d * D_STATE + n];
    const float wdt = w_dt[d], bdt = b_dt[d], Dv = Dp[d];

    float s[D_STATE];
    const size_t o = (((size_t)b * NCHUNK + c) * D_INNER + d) * D_STATE;
#pragma unroll
    for (int n = 0; n < D_STATE; ++n) s[n] = init[o + n];

    for (int tt = 0; tt < CHUNK; ++tt) {
        const int   t    = c * CHUNK + tt;
        const float uval = u[(size_t)(b * TLEN + t) * D_INNER + d];
        const float dt   = softplusf_(sxp[tt * NPAD] * wdt + bdt);
        const float du   = dt * uval;
        float yt = Dv * uval;
#pragma unroll
        for (int n = 0; n < D_STATE; ++n) {
            s[n] = s[n] * __expf(dt * Ad[n]) + du * sxp[tt * NPAD + 1 + n];
            yt  += s[n] * sxp[tt * NPAD + 1 + D_STATE + n];
        }
        y[(size_t)(b * TLEN + t) * D_INNER + d] = yt;
    }
}

// ---------------------------------------------------------------------------
extern "C" void kernel_launch(void* const* d_in, const int* in_sizes, int n_in,
                              void* d_out, int out_size, void* d_ws, size_t ws_size,
                              hipStream_t stream) {
    const float* x       = (const float*)d_in[0];
    const float* W_in    = (const float*)d_in[1];
    const float* conv_w  = (const float*)d_in[2];
    const float* conv_b  = (const float*)d_in[3];
    const float* W_xproj = (const float*)d_in[4];
    const float* w_dt    = (const float*)d_in[5];
    const float* b_dt    = (const float*)d_in[6];
    const float* A       = (const float*)d_in[7];
    const float* Dvec    = (const float*)d_in[8];
    const float* W_out   = (const float*)d_in[9];
    float*       out     = (float*)d_out;
    float*       ws      = (float*)d_ws;

    // f32 workspace layout; xz region reused for y once u exists.
    float* xz   = ws;                                                  // ROWS*1024
    float* u    = xz   + (size_t)ROWS * 2 * D_INNER;                   // ROWS*512
    float* xp   = u    + (size_t)ROWS * D_INNER;                       // ROWS*48
    float* wxp  = xp   + (size_t)ROWS * NPAD;                          // 512*48
    float* P    = wxp  + (size_t)D_INNER * NPAD;
    float* S    = P    + (size_t)BATCH * NCHUNK * D_INNER * D_STATE;
    float* init = S    + (size_t)BATCH * NCHUNK * D_INNER * D_STATE;
    float* fend = init + (size_t)BATCH * NCHUNK * D_INNER * D_STATE;
    float* y    = xz;                                                  // reuse

    // f16 packed-B regions (32B aligned).
    _Float16* hbase = (_Float16*)((((uintptr_t)fend) + 31) & ~(uintptr_t)31);
    const size_t szWin  = (size_t)((2 * D_INNER) / 16) * (D_MODEL / 32) * 32 * 16; // 262144
    const size_t szWxp  = (size_t)(NPAD / 16) * (D_INNER / 32) * 32 * 16;          //  24576
    _Float16* BpWin  = hbase;
    _Float16* BpWxp  = BpWin + szWin;
    _Float16* BpWout = BpWxp + szWxp;

    const int thr = 256;  // 8 waves per block

    // Pack weights into WMMA fragment order.
    {
        const int tW = ((2 * D_INNER) / 16) * (D_MODEL / 32) * 32;   // 16384
        pack_b_f16<<<(tW + thr - 1) / thr, thr, 0, stream>>>(W_in, BpWin, 2 * D_INNER, D_MODEL);
        const int tO = (D_MODEL / 16) * (D_INNER / 32) * 32;         //  8192
        pack_b_f16<<<(tO + thr - 1) / thr, thr, 0, stream>>>(W_out, BpWout, D_MODEL, D_INNER);
    }
    pad_wxproj<<<(D_INNER * NPAD + thr - 1) / thr, thr, 0, stream>>>(W_xproj, wxp);
    {
        const int tX = (NPAD / 16) * (D_INNER / 32) * 32;            //  1536
        pack_b_f16<<<(tX + thr - 1) / thr, thr, 0, stream>>>(wxp, BpWxp, NPAD, D_INNER);
    }

    // GEMM1: xz = x @ W_in   (16384 x 1024 x 256), 2x4 tiles/wave.
    {
        const int waves = ((ROWS / 16) / 2) * (((2 * D_INNER) / 16) / 4);  // 8192
        gemm_wmma_packed<2, 4><<<(waves * 32 + thr - 1) / thr, thr, 0, stream>>>(
            x, BpWin, xz, ROWS, 2 * D_INNER, D_MODEL);
    }

    // conv + SiLU gate -> u
    conv_gate<<<ROWS, D_INNER, 0, stream>>>(xz, conv_w, conv_b, u);

    // GEMM2: xp = u @ wxp  (16384 x 48 x 512), 2x3 tiles/wave.
    {
        const int waves = ((ROWS / 16) / 2) * ((NPAD / 16) / 3);           //   512
        gemm_wmma_packed<2, 3><<<(waves * 32 + thr - 1) / thr, thr, 0, stream>>>(
            u, BpWxp, xp, ROWS, NPAD, D_INNER);
    }

    // chunked selective scan
    scan_partial<<<BATCH * NCHUNK * 2, thr, 0, stream>>>(xp, u, w_dt, b_dt, A, P, S);
    scan_carry<<<(BATCH * D_INNER + thr - 1) / thr, thr, 0, stream>>>(P, S, init);
    scan_final<<<BATCH * NCHUNK * 2, thr, 0, stream>>>(xp, u, w_dt, b_dt, A, Dvec, init, y);

    // GEMM3: out = y @ W_out  (16384 x 256 x 512), 2x4 tiles/wave.
    {
        const int waves = ((ROWS / 16) / 2) * ((D_MODEL / 16) / 4);        //  2048
        gemm_wmma_packed<2, 4><<<(waves * 32 + thr - 1) / thr, thr, 0, stream>>>(
            y, BpWout, out, ROWS, D_MODEL, D_INNER);
    }
}
